// FlashAttention_11570641895611
// MI455X (gfx1250) — compile-verified, matrix operand order checked by reading, not run on hardware
//
#include <hip/hip_runtime.h>
#include <math.h>

typedef __attribute__((ext_vector_type(16))) _Float16 v16h;
typedef __attribute__((ext_vector_type(8)))  _Float16 v8h;
typedef __attribute__((ext_vector_type(8)))  float    v8f;

#define B_SZ   2
#define H_SZ   16
#define N_SZ   4096
#define D_SZ   64
#define BR     128    // query rows per block (4 waves x 32)
#define BC     64     // key rows per iteration
#define NWAVE  4
#define PITCH  72     // LDS row pitch in halves (64 + 8 pad, 16B-aligned chunks)

#if __has_builtin(__builtin_amdgcn_exp2f)
#define EXP2F __builtin_amdgcn_exp2f
#else
#define EXP2F exp2f
#endif

// 16-bit operand fragment from an LDS row (CDNA5 A/B layout):
// lanes 0-15 take K=[kb..kb+7],[kb+16..kb+23]; lanes 16-31 take K=[kb+8..15],[kb+24..31].
__device__ __forceinline__ v16h frag_lds(const _Float16* row, int kb, int h) {
    v8h lo = *reinterpret_cast<const v8h*>(row + kb + 8 * h);
    v8h hi = *reinterpret_cast<const v8h*>(row + kb + 16 + 8 * h);
    return __builtin_shufflevector(lo, hi, 0,1,2,3,4,5,6,7,8,9,10,11,12,13,14,15);
}

// Same fragment pattern from a global f32 row, scaled and narrowed to f16.
__device__ __forceinline__ v16h qfrag(const float* row, int kb, int h, float scale) {
    float4 a0 = *reinterpret_cast<const float4*>(row + kb + 8 * h);
    float4 a1 = *reinterpret_cast<const float4*>(row + kb + 8 * h + 4);
    float4 b0 = *reinterpret_cast<const float4*>(row + kb + 16 + 8 * h);
    float4 b1 = *reinterpret_cast<const float4*>(row + kb + 16 + 8 * h + 4);
    v16h o;
    o[0]=(_Float16)(a0.x*scale); o[1]=(_Float16)(a0.y*scale);
    o[2]=(_Float16)(a0.z*scale); o[3]=(_Float16)(a0.w*scale);
    o[4]=(_Float16)(a1.x*scale); o[5]=(_Float16)(a1.y*scale);
    o[6]=(_Float16)(a1.z*scale); o[7]=(_Float16)(a1.w*scale);
    o[8]=(_Float16)(b0.x*scale); o[9]=(_Float16)(b0.y*scale);
    o[10]=(_Float16)(b0.z*scale); o[11]=(_Float16)(b0.w*scale);
    o[12]=(_Float16)(b1.x*scale); o[13]=(_Float16)(b1.y*scale);
    o[14]=(_Float16)(b1.z*scale); o[15]=(_Float16)(b1.w*scale);
    return o;
}

// Cross-lane combine with lane^16 via ds_swizzle (group-of-32, xor=16).
__device__ __forceinline__ float xmax16(float x) {
    int t = __builtin_amdgcn_ds_swizzle(__float_as_int(x), 0x401f);
    return fmaxf(x, __int_as_float(t));
}
__device__ __forceinline__ float xadd16(float x) {
    int t = __builtin_amdgcn_ds_swizzle(__float_as_int(x), 0x401f);
    return x + __int_as_float(t);
}

// Online-softmax update for one 16-query column tile (logits in base-2 domain).
__device__ __forceinline__ void softmax_tile(v8f s[4], float& m_run, float& l_run,
                                             v8f* acc4, _Float16* prow, int h) {
    float pm = -INFINITY;
#pragma unroll
    for (int ct = 0; ct < 4; ++ct)
#pragma unroll
        for (int r = 0; r < 8; ++r) pm = fmaxf(pm, s[ct][r]);
    const float m_new = fmaxf(m_run, xmax16(pm));
    const float alpha = EXP2F(m_run - m_new);
    float rs = 0.0f;
#pragma unroll
    for (int ct = 0; ct < 4; ++ct) {
        v8h pf;
#pragma unroll
        for (int r = 0; r < 8; ++r) {
            float p = EXP2F(s[ct][r] - m_new);
            pf[r] = (_Float16)p;
            rs += p;
        }
        *reinterpret_cast<v8h*>(prow + ct * 16 + 8 * h) = pf;
    }
    rs = xadd16(rs);
    l_run = alpha * l_run + rs;
    m_run = m_new;
#pragma unroll
    for (int dt = 0; dt < 4; ++dt)
#pragma unroll
        for (int r = 0; r < 8; ++r) acc4[dt][r] *= alpha;
}

__global__ __launch_bounds__(128) void fa_fwd_f16wmma_db(
        const float* __restrict__ q, const float* __restrict__ k,
        const float* __restrict__ v, float* __restrict__ out) {
    // Double-buffered K/V tiles; per-wave P scratch.
    __shared__ __align__(16) _Float16 Kl[2][BC][PITCH];        // K tile [key c][d]
    __shared__ __align__(16) _Float16 Vt[2][D_SZ][PITCH];      // V^T tile [d][key c]
    __shared__ __align__(16) _Float16 Pq[NWAVE][32][PITCH];    // P^T rows per query [q][c]

    const int tid  = threadIdx.x;
    const int wave = tid >> 5;
    const int lane = tid & 31;
    const int l16  = lane & 15;
    const int h    = lane >> 4;

    const int bh   = blockIdx.y;
    const int row0 = blockIdx.x * BR;
    const size_t base = (size_t)bh * N_SZ * D_SZ;

    // Reference logits = (q.k)/D; fold 1/D AND log2(e) -> softmax in exp2 domain.
    const float scale = 1.4426950408889634f / (float)D_SZ;

    const int   qgA = row0 + wave * 32 + l16;
    const int   qgB = qgA + 16;
    const float* qrwA = q + base + (size_t)qgA * D_SZ;
    const float* qrwB = q + base + (size_t)qgB * D_SZ;
    const v16h bqA0 = qfrag(qrwA, 0,  h, scale);
    const v16h bqA1 = qfrag(qrwA, 32, h, scale);
    const v16h bqB0 = qfrag(qrwB, 0,  h, scale);
    const v16h bqB1 = qfrag(qrwB, 32, h, scale);

    v8f acc[8] = {};   // O^T accumulators: [0..3] tile A, [4..7] tile B
    float mA = -INFINITY, lA = 0.0f, mB = -INFINITY, lB = 0.0f;

    // Tile-fill assignment: each thread owns half a row (32 f32) of K and V.
    const int c   = tid >> 1;
    const int dh2 = (tid & 1) * 32;

    // Tile fill into buffer `buf` for key block at `kb` (+ prefetch the next tile at kb + BC).
    auto load_tile = [&](int buf, int kb) {
        const float* krow = k + base + (size_t)(kb + c) * D_SZ + dh2;
        const float* vrow = v + base + (size_t)(kb + c) * D_SZ + dh2;
#pragma unroll
        for (int j = 0; j < 32; j += 8) {
            float4 ka = *reinterpret_cast<const float4*>(krow + j);
            float4 kc = *reinterpret_cast<const float4*>(krow + j + 4);
            v8h kh;
            kh[0]=(_Float16)ka.x; kh[1]=(_Float16)ka.y; kh[2]=(_Float16)ka.z; kh[3]=(_Float16)ka.w;
            kh[4]=(_Float16)kc.x; kh[5]=(_Float16)kc.y; kh[6]=(_Float16)kc.z; kh[7]=(_Float16)kc.w;
            *reinterpret_cast<v8h*>(&Kl[buf][c][dh2 + j]) = kh;
            float4 va = *reinterpret_cast<const float4*>(vrow + j);
            float4 vc = *reinterpret_cast<const float4*>(vrow + j + 4);
            Vt[buf][dh2 + j + 0][c] = (_Float16)va.x;  Vt[buf][dh2 + j + 1][c] = (_Float16)va.y;
            Vt[buf][dh2 + j + 2][c] = (_Float16)va.z;  Vt[buf][dh2 + j + 3][c] = (_Float16)va.w;
            Vt[buf][dh2 + j + 4][c] = (_Float16)vc.x;  Vt[buf][dh2 + j + 5][c] = (_Float16)vc.y;
            Vt[buf][dh2 + j + 6][c] = (_Float16)vc.z;  Vt[buf][dh2 + j + 7][c] = (_Float16)vc.w;
        }
        if (kb + BC < N_SZ) {
            __builtin_prefetch(krow + (size_t)BC * D_SZ, 0, 1);  // global_prefetch_b8
            __builtin_prefetch(vrow + (size_t)BC * D_SZ, 0, 1);
        }
    };

    load_tile(0, 0);   // prologue fill
    int cur = 0;

    for (int kb = 0; kb < N_SZ; kb += BC) {
        __syncthreads();   // buffer `cur` (written last iteration / prologue) is ready

        // ---- S^T = K * Q^T for both query tiles; A (K rows) loaded once, used twice ----
        v8f s0[4], s1[4];
#pragma unroll
        for (int ct = 0; ct < 4; ++ct) {
            const _Float16* krl = &Kl[cur][ct * 16 + l16][0];
            v16h ak0 = frag_lds(krl, 0,  h);
            v16h ak1 = frag_lds(krl, 32, h);
            v8f z0 = {}, z1 = {};
            z0 = __builtin_amdgcn_wmma_f32_16x16x32_f16(false, ak0, false, bqA0, (short)0, z0, false, false);
            z1 = __builtin_amdgcn_wmma_f32_16x16x32_f16(false, ak0, false, bqB0, (short)0, z1, false, false);
            z0 = __builtin_amdgcn_wmma_f32_16x16x32_f16(false, ak1, false, bqA1, (short)0, z0, false, false);
            z1 = __builtin_amdgcn_wmma_f32_16x16x32_f16(false, ak1, false, bqB1, (short)0, z1, false, false);
            s0[ct] = z0; s1[ct] = z1;
        }

        // ---- Online softmax per tile (base-2 domain) ----
        softmax_tile(s0, mA, lA, &acc[0], &Pq[wave][l16][0],      h);
        softmax_tile(s1, mB, lB, &acc[4], &Pq[wave][16 + l16][0], h);

        // ---- Overlap: fill the OTHER buffer with the next K/V tile mid-iteration ----
        if (kb + BC < N_SZ) load_tile(cur ^ 1, kb + BC);

        // ---- O^T += V^T * P^T; A (V^T rows) loaded once, used twice ----
        const _Float16* prow0 = &Pq[wave][l16][0];
        const _Float16* prow1 = &Pq[wave][16 + l16][0];
        v16h bpA0 = frag_lds(prow0, 0,  h);
        v16h bpA1 = frag_lds(prow0, 32, h);
        v16h bpB0 = frag_lds(prow1, 0,  h);
        v16h bpB1 = frag_lds(prow1, 32, h);
#pragma unroll
        for (int dt = 0; dt < 4; ++dt) {
            const _Float16* vrl = &Vt[cur][dt * 16 + l16][0];
            v16h av0 = frag_lds(vrl, 0,  h);
            v16h av1 = frag_lds(vrl, 32, h);
            acc[dt]     = __builtin_amdgcn_wmma_f32_16x16x32_f16(false, av0, false, bpA0, (short)0, acc[dt],     false, false);
            acc[4 + dt] = __builtin_amdgcn_wmma_f32_16x16x32_f16(false, av0, false, bpB0, (short)0, acc[4 + dt], false, false);
            acc[dt]     = __builtin_amdgcn_wmma_f32_16x16x32_f16(false, av1, false, bpA1, (short)0, acc[dt],     false, false);
            acc[4 + dt] = __builtin_amdgcn_wmma_f32_16x16x32_f16(false, av1, false, bpB1, (short)0, acc[4 + dt], false, false);
        }
        cur ^= 1;
    }

    // ---- Epilogue: consecutive r = consecutive d -> float4 stores ----
    const float invA = 1.0f / fmaxf(lA, 1e-6f);
    const float invB = 1.0f / fmaxf(lB, 1e-6f);
    float* orowA = out + base + (size_t)qgA * D_SZ;
    float* orowB = out + base + (size_t)qgB * D_SZ;
#pragma unroll
    for (int dt = 0; dt < 4; ++dt) {
        float4 o0, o1;
        o0.x = acc[dt][0] * invA; o0.y = acc[dt][1] * invA;
        o0.z = acc[dt][2] * invA; o0.w = acc[dt][3] * invA;
        o1.x = acc[dt][4] * invA; o1.y = acc[dt][5] * invA;
        o1.z = acc[dt][6] * invA; o1.w = acc[dt][7] * invA;
        *reinterpret_cast<float4*>(orowA + dt * 16 + 8 * h)     = o0;
        *reinterpret_cast<float4*>(orowA + dt * 16 + 8 * h + 4) = o1;
        float4 p0, p1;
        p0.x = acc[4+dt][0] * invB; p0.y = acc[4+dt][1] * invB;
        p0.z = acc[4+dt][2] * invB; p0.w = acc[4+dt][3] * invB;
        p1.x = acc[4+dt][4] * invB; p1.y = acc[4+dt][5] * invB;
        p1.z = acc[4+dt][6] * invB; p1.w = acc[4+dt][7] * invB;
        *reinterpret_cast<float4*>(orowB + dt * 16 + 8 * h)     = p0;
        *reinterpret_cast<float4*>(orowB + dt * 16 + 8 * h + 4) = p1;
    }
}

extern "C" void kernel_launch(void* const* d_in, const int* in_sizes, int n_in,
                              void* d_out, int out_size, void* d_ws, size_t ws_size,
                              hipStream_t stream) {
    (void)in_sizes; (void)n_in; (void)d_ws; (void)ws_size; (void)out_size;
    const float* q = (const float*)d_in[0];
    const float* k = (const float*)d_in[1];
    const float* v = (const float*)d_in[2];
    float* out = (float*)d_out;

    dim3 grid(N_SZ / BR, B_SZ * H_SZ);   // 32 x 32 blocks
    dim3 block(128);                      // 4 wave32 waves
    fa_fwd_f16wmma_db<<<grid, block, 0, stream>>>(q, k, v, out);
}
